// LDGNNetwork_67886253081261
// MI455X (gfx1250) — matrix operand order repaired and checked by reading, not compiled, not run on hardware
//
#include <hip/hip_runtime.h>
#include <hip/hip_bf16.h>

#define N_NODES 32768
#define N_EDGES 524288
#define BSZ 512
#define AG 64          // nodes per graph
#define DEG 16
#define F_IN 32
#define NH 4
#define CC 64
#define HC 256
#define NOUT 5
#define LATW 576       // CC + 2*HC

typedef __attribute__((ext_vector_type(16))) _Float16 v16h;
typedef __attribute__((ext_vector_type(8)))  float    v8f;

__device__ __forceinline__ v8f wmma_f16(v16h a, v16h b, v8f c) {
  return __builtin_amdgcn_wmma_f32_16x16x32_f16(false, a, false, b, (short)0, c, false, false);
}

// ---- A fragment from f32 global (ISA 7.12.2 16-bit A 16x32 layout) -------
// lanes 0-15: M=lane, K=k0+0..7 (v0-3), k0+16..23 (v4-7); lanes 16-31: K+8.
__device__ __forceinline__ v16h load_a_frag_g(const float* __restrict__ A, int lda,
                                              int row_base, int k0,
                                              const float* __restrict__ rowscale) {
  int lane = threadIdx.x & 31;
  int m = lane & 15, hf = lane >> 4;
  int row = row_base + m;
  float s = rowscale ? rowscale[row] : 1.0f;
  const float4* q  = (const float4*)(A + (size_t)row * lda + k0 + 8 * hf);
  const float4* q2 = (const float4*)(A + (size_t)row * lda + k0 + 16 + 8 * hf);
  float4 f0 = q[0], f1 = q[1], f2 = q2[0], f3 = q2[1];
  v16h a;
  a[0]  = (_Float16)(f0.x * s); a[1]  = (_Float16)(f0.y * s);
  a[2]  = (_Float16)(f0.z * s); a[3]  = (_Float16)(f0.w * s);
  a[4]  = (_Float16)(f1.x * s); a[5]  = (_Float16)(f1.y * s);
  a[6]  = (_Float16)(f1.z * s); a[7]  = (_Float16)(f1.w * s);
  a[8]  = (_Float16)(f2.x * s); a[9]  = (_Float16)(f2.y * s);
  a[10] = (_Float16)(f2.z * s); a[11] = (_Float16)(f2.w * s);
  a[12] = (_Float16)(f3.x * s); a[13] = (_Float16)(f3.y * s);
  a[14] = (_Float16)(f3.z * s); a[15] = (_Float16)(f3.w * s);
  return a;
}

__device__ __forceinline__ v16h load_a_frag_lds(const _Float16* T, int ldt, int k0) {
  int lane = threadIdx.x & 31;
  int m = lane & 15, hf = lane >> 4;
  const _Float16* p  = T + m * ldt + k0 + 8 * hf;
  const _Float16* p2 = T + m * ldt + k0 + 16 + 8 * hf;
  v16h a;
#pragma unroll
  for (int i = 0; i < 8; i++) a[i] = p[i];
#pragma unroll
  for (int i = 0; i < 8; i++) a[8 + i] = p2[i];
  return a;
}

// ---- packed-f16 B fragments: one contiguous 32B load per lane ------------
// layout: [kt][nt][lane 0..31][16 half]; built once by pack_b_kernel.
__device__ __forceinline__ v16h load_b_frag_packed(const _Float16* __restrict__ P,
                                                   int NT, int kt, int nt) {
  int lane = threadIdx.x & 31;
  return *(const v16h*)(P + (((size_t)kt * NT + nt) * 32 + lane) * 16);
}

__global__ void pack_b_kernel(const float* __restrict__ B, int K, int N,
                              _Float16* __restrict__ P) {
  int t = blockIdx.x * blockDim.x + threadIdx.x;
  int total = (K / 32) * (N / 16) * 32;
  if (t >= total) return;
  int lane = t & 31;
  int f = t >> 5;
  int NT = N / 16;
  int kt = f / NT, nt = f - kt * NT;
  int n = nt * 16 + (lane & 15);
  int hf = lane >> 4;
  _Float16* dst = P + (size_t)f * 512 + lane * 16;
#pragma unroll
  for (int r = 0; r < 8; r++) {
    int k = kt * 32 + 8 * hf + 2 * r + (r >= 4 ? 8 : 0);
    dst[2 * r]     = (_Float16)B[(size_t)k * N + n];
    dst[2 * r + 1] = (_Float16)B[(size_t)(k + 1) * N + n];
  }
}

// ---- monotone float<->uint encoding for atomicMax-based segment max ------
__device__ __forceinline__ unsigned fenc(float f) {
  unsigned u = __float_as_uint(f);
  return (u & 0x80000000u) ? ~u : (u | 0x80000000u);
}
__device__ __forceinline__ float fdec(unsigned k) {
  unsigned u = (k & 0x80000000u) ? (k & 0x7FFFFFFFu) : ~k;
  return __uint_as_float(u);
}

// ===== encoder: h = relu(relu(x@W1+b1)@W2+b2), fused via LDS f16 tile =====
__global__ void encoder_kernel(const float* __restrict__ x,
                               const _Float16* __restrict__ pW1, const float* __restrict__ b1,
                               const _Float16* __restrict__ pW2, const float* __restrict__ b2,
                               float* __restrict__ h) {
  __shared__ _Float16 t[16 * CC];
  const int wave = threadIdx.x >> 5;
  const int lane = threadIdx.x & 31;
  const int row_base = blockIdx.x << 4;
  const int nt = wave;            // 4 waves x 16 cols = 64, NT = 4
  const int n = (nt << 4) + (lane & 15);
  const int hf = lane >> 4;

  // stage 1: 16x32 @ 32x64 (K=32 -> one WMMA per col tile)
  v16h a = load_a_frag_g(x, F_IN, row_base, 0, nullptr);
  v16h b = load_b_frag_packed(pW1, 4, 0, nt);
  v8f c = {};
  c = wmma_f16(a, b, c);
  float bv = b1[n];
#pragma unroll
  for (int r = 0; r < 8; r++) {
    float v = c[r] + bv;
    v = v > 0.f ? v : 0.f;
    t[(8 * hf + r) * CC + n] = (_Float16)v;
  }
  __syncthreads();

  // stage 2: 16x64 @ 64x64 (K=64 -> two WMMA, A frags from LDS)
  v8f c2 = {};
#pragma unroll
  for (int kt = 0; kt < 2; kt++) {
    v16h a2 = load_a_frag_lds(t, CC, kt * 32);
    v16h bf2 = load_b_frag_packed(pW2, 4, kt, nt);
    c2 = wmma_f16(a2, bf2, c2);
  }
  float b2s = b2[n];
#pragma unroll
  for (int r = 0; r < 8; r++) {
    float v = c2[r] + b2s;
    v = v > 0.f ? v : 0.f;
    h[(size_t)(row_base + 8 * hf + r) * CC + n] = v;
  }
}

// ===== dual GEMM: DL = A@BL+bl, DR = A@BR+br (shared A fragments) =========
__global__ void dual_gemm_kernel(const float* __restrict__ A, int K,
                                 const _Float16* __restrict__ PL, const float* __restrict__ biasL,
                                 const _Float16* __restrict__ PR, const float* __restrict__ biasR,
                                 float* __restrict__ DL, float* __restrict__ DR,
                                 const float* __restrict__ rowscale) {
  const int wave = threadIdx.x >> 5;
  const int lane = threadIdx.x & 31;
  const int row_base = blockIdx.x << 4;
  const int nt = (blockIdx.y << 2) + wave;   // NT = HC/16 = 16
  v8f cl = {}, cr = {};
  const int KT = K >> 5;
  for (int kt = 0; kt < KT; kt++) {
    if (kt + 1 < KT)
      __builtin_prefetch(A + (size_t)(row_base + (lane & 15)) * K + kt * 32 + 32, 0, 0);
    v16h a  = load_a_frag_g(A, K, row_base, kt * 32, rowscale);
    v16h bl = load_b_frag_packed(PL, 16, kt, nt);
    v16h br = load_b_frag_packed(PR, 16, kt, nt);
    cl = wmma_f16(a, bl, cl);
    cr = wmma_f16(a, br, cr);
  }
  const int n = (nt << 4) + (lane & 15);
  const int hf = lane >> 4;
  float bvl = biasL[n], bvr = biasR[n];
#pragma unroll
  for (int r = 0; r < 8; r++) {
    size_t idx = (size_t)(row_base + 8 * hf + r) * HC + n;
    DL[idx] = cl[r] + bvl;
    DR[idx] = cr[r] + bvr;
  }
}

// ===== per-graph GATv2 edge kernel: whole scatter-softmax in LDS ==========
// LDS: xl tile 64x256, xr tile 64x256 (reused as accumulator), We 3x256,
//      att 256, max/den 64x4 each  ->  ~134 KB (< 320 KB/WGP on CDNA5)
#define SMEM_GAT ((2 * AG * HC + 3 * HC + HC + 2 * AG * NH) * 4)

__global__ void gat_edge_kernel(const float* __restrict__ xl, const float* __restrict__ xr,
                                const float* __restrict__ We, const float* __restrict__ att,
                                const float* __restrict__ bo,
                                const int* __restrict__ esrc, const int* __restrict__ edst,
                                const float* __restrict__ eattr,
                                float* __restrict__ out) {
  extern __shared__ float sm[];
  float*    xlb  = sm;               // 64*256
  float*    xrb  = xlb + AG * HC;    // 64*256 (becomes acc after pass 1)
  float*    web  = xrb + AG * HC;    // 3*256
  float*    attb = web + 3 * HC;     // 256
  unsigned* mxb  = (unsigned*)(attb + HC);   // 64*4
  float*    denb = (float*)(mxb + AG * NH);  // 64*4

  const int g = blockIdx.x;
  const int tid = threadIdx.x;
  const int nodebase = g * AG;
  const int ebase = g * AG * DEG;

  // async staging: global -> LDS via CDNA5 ASYNCcnt path (16B per lane)
  const float4* xl4 = (const float4*)(xl + (size_t)nodebase * HC);
  const float4* xr4 = (const float4*)(xr + (size_t)nodebase * HC);
  float4* xlb4 = (float4*)xlb;
  float4* xrb4 = (float4*)xrb;
  for (int i = tid; i < AG * HC / 4; i += 256) {
    unsigned l0 = (unsigned)(size_t)(xlb4 + i);   // LDS byte offset = addr[31:0]
    unsigned l1 = (unsigned)(size_t)(xrb4 + i);
    unsigned long long g0 = (unsigned long long)(size_t)(xl4 + i);
    unsigned long long g1 = (unsigned long long)(size_t)(xr4 + i);
    asm volatile("global_load_async_to_lds_b128 %0, %1, off"
                 :: "v"(l0), "v"(g0) : "memory");
    asm volatile("global_load_async_to_lds_b128 %0, %1, off"
                 :: "v"(l1), "v"(g1) : "memory");
  }
  for (int i = tid; i < 3 * HC; i += 256) web[i] = We[i];
  for (int i = tid; i < HC; i += 256) attb[i] = att[i];
  for (int i = tid; i < AG * NH; i += 256) { mxb[i] = 0u; denb[i] = 0.f; }
  asm volatile("s_wait_asynccnt 0x0" ::: "memory");
  __syncthreads();

  int src[4], dst[4];
  float lg[4][NH];

  // pass 1: logits + segment max (LDS atomicMax on monotone-encoded floats)
  for (int k = 0; k < 4; k++) {
    int el = tid + k * 256;
    int e = ebase + el;
    int s = esrc[e] - nodebase;
    int d = edst[e] - nodebase;
    src[k] = s; dst[k] = d;
    float a0 = eattr[(size_t)e * 3 + 0];
    float a1 = eattr[(size_t)e * 3 + 1];
    float a2 = eattr[(size_t)e * 3 + 2];
#pragma unroll
    for (int hh = 0; hh < NH; hh++) {
      const float* xls = xlb + s * HC + hh * CC;
      const float* xrd = xrb + d * HC + hh * CC;
      const float* w0 = web + hh * CC;
      const float* w1 = web + HC + hh * CC;
      const float* w2 = web + 2 * HC + hh * CC;
      const float* at = attb + hh * CC;
      float acc = 0.f;
#pragma unroll 8
      for (int c = 0; c < CC; c++) {
        float sv = xls[c] + xrd[c] + a0 * w0[c] + a1 * w1[c] + a2 * w2[c];
        sv = sv > 0.f ? sv : 0.2f * sv;   // leaky_relu(0.2)
        acc += sv * at[c];
      }
      lg[k][hh] = acc;
      atomicMax(&mxb[d * NH + hh], fenc(acc));
    }
  }
  __syncthreads();

  // zero accumulator (xrb no longer needed) + pass 2: exp & denominators
  for (int i = tid; i < AG * HC; i += 256) xrb[i] = 0.f;
  float ex[4][NH];
  for (int k = 0; k < 4; k++) {
#pragma unroll
    for (int hh = 0; hh < NH; hh++) {
      float m = fdec(mxb[dst[k] * NH + hh]);
      float v = __expf(lg[k][hh] - m);
      ex[k][hh] = v;
      atomicAdd(&denb[dst[k] * NH + hh], v);
    }
  }
  __syncthreads();

  // pass 3: out[dst] += alpha * xl[src]  (LDS ds_add_f32)
  for (int k = 0; k < 4; k++) {
#pragma unroll
    for (int hh = 0; hh < NH; hh++) {
      float alpha = ex[k][hh] / (denb[dst[k] * NH + hh] + 1e-16f);
      float* accp = xrb + dst[k] * HC + hh * CC;
      const float* xls = xlb + src[k] * HC + hh * CC;
#pragma unroll 8
      for (int c = 0; c < CC; c++) atomicAdd(&accp[c], alpha * xls[c]);
    }
  }
  __syncthreads();

  // epilogue: + bo, relu (layer output relu), write back
  for (int i = tid; i < AG * HC; i += 256) {
    int ch = i & (HC - 1);
    float v = xrb[i] + bo[ch];
    v = v > 0.f ? v : 0.f;
    out[(size_t)nodebase * HC + i] = v;
  }
}

// ===== gathers + final head ==============================================
__global__ void gather_kernel(const int* __restrict__ ctrl, const float* __restrict__ src,
                              int width, float* __restrict__ lat, int off) {
  int i = blockIdx.x * blockDim.x + threadIdx.x;
  if (i >= BSZ * width) return;
  int b = i / width, j = i - b * width;
  lat[(size_t)b * LATW + off + j] = src[(size_t)ctrl[b] * width + j];
}

__global__ void final_kernel(const float* __restrict__ lat, const float* __restrict__ Wf,
                             const float* __restrict__ bf, float* __restrict__ out) {
  int b = blockIdx.x * blockDim.x + threadIdx.x;
  if (b >= BSZ) return;
  float acc[NOUT];
#pragma unroll
  for (int o = 0; o < NOUT; o++) acc[o] = bf[o];
  const float* row = lat + (size_t)b * LATW;
  for (int k = 0; k < LATW; k++) {
    float v = row[k];
#pragma unroll
    for (int o = 0; o < NOUT; o++) acc[o] += v * Wf[k * NOUT + o];
  }
#pragma unroll
  for (int o = 0; o < NOUT; o++) out[b * NOUT + o] = acc[o];
}

extern "C" void kernel_launch(void* const* d_in, const int* in_sizes, int n_in,
                              void* d_out, int out_size, void* d_ws, size_t ws_size,
                              hipStream_t stream) {
  (void)in_sizes; (void)n_in; (void)out_size; (void)ws_size;
  const float* x       = (const float*)d_in[0];
  const float* eattr   = (const float*)d_in[1];
  const float* dm_mask = (const float*)d_in[2];
  const float* W_enc1  = (const float*)d_in[3];
  const float* b_enc1  = (const float*)d_in[4];
  const float* W_enc2  = (const float*)d_in[5];
  const float* b_enc2  = (const float*)d_in[6];
  const float* Wl1 = (const float*)d_in[7];
  const float* bl1 = (const float*)d_in[8];
  const float* Wr1 = (const float*)d_in[9];
  const float* br1 = (const float*)d_in[10];
  const float* We1 = (const float*)d_in[11];
  const float* att1= (const float*)d_in[12];
  const float* bo1 = (const float*)d_in[13];
  const float* Wl2 = (const float*)d_in[14];
  const float* bl2 = (const float*)d_in[15];
  const float* Wr2 = (const float*)d_in[16];
  const float* br2 = (const float*)d_in[17];
  const float* We2 = (const float*)d_in[18];
  const float* att2= (const float*)d_in[19];
  const float* bo2 = (const float*)d_in[20];
  const float* Wf  = (const float*)d_in[21];
  const float* bf  = (const float*)d_in[22];
  const int* esrc  = (const int*)d_in[23];
  const int* edst  = (const int*)d_in[24];
  const int* ctrl  = (const int*)d_in[25];
  float* out = (float*)d_out;
  float* ws  = (float*)d_ws;

  // workspace layout (floats): h | xl | xr | h1 | h2 | lat | packed f16 weights
  float* h   = ws;
  float* xl  = h  + (size_t)N_NODES * CC;
  float* xr  = xl + (size_t)N_NODES * HC;
  float* h1  = xr + (size_t)N_NODES * HC;
  float* h2  = h1 + (size_t)N_NODES * HC;
  float* lat = h2 + (size_t)N_NODES * HC;
  _Float16* pWe1 = (_Float16*)(lat + (size_t)BSZ * LATW);
  _Float16* pWe2 = pWe1 + 4 * 512;     // W_enc1: 1x4 frags
  _Float16* pWl1 = pWe2 + 8 * 512;     // W_enc2: 2x4 frags
  _Float16* pWr1 = pWl1 + 32 * 512;    // Wl1/Wr1: 2x16 frags
  _Float16* pWl2 = pWr1 + 32 * 512;
  _Float16* pWr2 = pWl2 + 128 * 512;   // Wl2/Wr2: 8x16 frags

  // one-time (per launch) weight packing into wave32 WMMA fragment layout
  pack_b_kernel<<<1, 128, 0, stream>>>(W_enc1, 32, 64, pWe1);
  pack_b_kernel<<<1, 256, 0, stream>>>(W_enc2, 64, 64, pWe2);
  pack_b_kernel<<<4, 256, 0, stream>>>(Wl1, 64, HC, pWl1);
  pack_b_kernel<<<4, 256, 0, stream>>>(Wr1, 64, HC, pWr1);
  pack_b_kernel<<<16, 256, 0, stream>>>(Wl2, HC, HC, pWl2);
  pack_b_kernel<<<16, 256, 0, stream>>>(Wr2, HC, HC, pWr2);

  // encoder MLP (WMMA, fused 2 layers)
  encoder_kernel<<<N_NODES / 16, 128, 0, stream>>>(x, pWe1, b_enc1, pWe2, b_enc2, h);
  gather_kernel<<<(BSZ * CC + 255) / 256, 256, 0, stream>>>(ctrl, h, CC, lat, 0);

  // layer 1: xl/xr transforms (dual WMMA GEMM), then per-graph LDS GAT
  dual_gemm_kernel<<<dim3(N_NODES / 16, HC / 64), 128, 0, stream>>>(
      h, CC, pWl1, bl1, pWr1, br1, xl, xr, nullptr);
  gat_edge_kernel<<<BSZ, 256, SMEM_GAT, stream>>>(xl, xr, We1, att1, bo1, esrc, edst, eattr, h1);
  gather_kernel<<<(BSZ * HC + 255) / 256, 256, 0, stream>>>(ctrl, h1, HC, lat, CC);

  // layer 2: dm_mask fused as A-row scale (x2 gathered from pre-mask h1 above)
  dual_gemm_kernel<<<dim3(N_NODES / 16, HC / 64), 128, 0, stream>>>(
      h1, HC, pWl2, bl2, pWr2, br2, xl, xr, dm_mask);
  gat_edge_kernel<<<BSZ, 256, SMEM_GAT, stream>>>(xl, xr, We2, att2, bo2, esrc, edst, eattr, h2);
  gather_kernel<<<(BSZ * HC + 255) / 256, 256, 0, stream>>>(ctrl, h2, HC, lat, CC + HC);

  // readout head
  final_kernel<<<(BSZ + 127) / 128, 128, 0, stream>>>(lat, Wf, bf, out);
}
